// LSTMlayer_39067022524724
// MI455X (gfx1250) — compile-verified, matrix operand order checked by reading, not run on hardware
//
#include <hip/hip_runtime.h>
#include <hip/hip_bf16.h>

// ---------------------------------------------------------------------------
// BatchNorm-LSTM, T=2048 B=32 I=H=256, fp32 in/out.
//   K0: split-convert X, W_ih, W_hh to bf16 hi/lo pairs (bf16x3 precision trick)
//   K1: bulk WMMA GEMM  a_ih = X @ W_ih^T  (16x64 strip per wave, A-reuse x4)
//   K2: per-timestep mean / rsqrt(var+eps) of a_ih  (scalar-BN stats)
//   K3: persistent single-workgroup recurrence (32 wave32 waves, 1 WGP):
//       per step: z = h @ W_hh^T via v_wmma_f32_16x16x32_bf16 (x3 split),
//       scalar-BN via shuffles+LDS, gates in-register, channel-BN via
//       ds_add_f32 atomics, h -> LDS bf16; next-step a_ih prefetched with
//       global_prefetch_b8 to hide HBM latency behind the WMMA chain.
// ---------------------------------------------------------------------------

typedef __bf16 bf16_t;
typedef __attribute__((ext_vector_type(16))) __bf16 v16bf;
typedef __attribute__((ext_vector_type(8)))  __bf16 v8bf;
typedef __attribute__((ext_vector_type(8)))  float  v8f;

#define CAT16(a,b) (__builtin_shufflevector((a),(b),0,1,2,3,4,5,6,7,8,9,10,11,12,13,14,15))
#define WMMA_BF16(A,B,C) __builtin_amdgcn_wmma_f32_16x16x32_bf16(false,(A),false,(B),(short)0,(C),false,false)

#define T_STEPS 2048
#define BATCH   32
#define HID     256
#define GATES   1024   // 4*HID
#define EPS     1e-5f

static __device__ __forceinline__ float fsigmoid(float x){ return 1.0f/(1.0f+__expf(-x)); }
static __device__ __forceinline__ float ftanh(float x){
  float xc = fminf(fmaxf(x,-15.0f),15.0f);
  float e  = __expf(2.0f*xc);
  return (e-1.0f)/(e+1.0f);
}

// ---------------------------------------------------------------- K0: split-convert
__global__ void cvt_split_kernel(const float* __restrict__ s,
                                 bf16_t* __restrict__ hi,
                                 bf16_t* __restrict__ lo, int n){
  int i = blockIdx.x*256 + threadIdx.x;
  if (i < n){
    float x = s[i];
    bf16_t h = (bf16_t)x;
    hi[i] = h;
    lo[i] = (bf16_t)(x - (float)h);
  }
}

// ---------------------------------------------------------------- K1: bulk GEMM a_ih
// One wave computes a 16x64 strip (4 adjacent 16x16 tiles sharing one A frag).
// A rows from X (row-major, K contiguous); B "columns" are rows of W_ih
// (W stored [N,K] row-major == column-major KxN, exactly the WMMA B layout).
__global__ void __launch_bounds__(256) gemm_ih_kernel(
    const bf16_t* __restrict__ Xh, const bf16_t* __restrict__ Xl,
    const bf16_t* __restrict__ Wh, const bf16_t* __restrict__ Wl,
    float* __restrict__ out){
  int w    = blockIdx.x*8 + (threadIdx.x >> 5);   // 0..65535
  int lane = threadIdx.x & 31;
  int ln   = lane & 15;
  int mt   = w >> 4;          // 0..4095 row tiles of 16 over T*B
  int ntg  = w & 15;          // group of 4 col tiles over 4H
  int khalf = (lane < 16) ? 0 : 8;   // 16-bit A/B fragment K layout

  const bf16_t* xph = Xh + (size_t)(mt*16 + ln)*256;
  const bf16_t* xpl = Xl + (size_t)(mt*16 + ln)*256;

  v8f acc[4] = {{},{},{},{}};
  #pragma unroll
  for (int ks = 0; ks < 8; ++ks){
    int kb = ks*32 + khalf;
    v16bf Ah = CAT16(*(const v8bf*)(xph+kb), *(const v8bf*)(xph+kb+16));
    v16bf Al = CAT16(*(const v8bf*)(xpl+kb), *(const v8bf*)(xpl+kb+16));
    #pragma unroll
    for (int n = 0; n < 4; ++n){
      const bf16_t* wph = Wh + (size_t)((ntg*4+n)*16 + ln)*256 + kb;
      const bf16_t* wpl = Wl + (size_t)((ntg*4+n)*16 + ln)*256 + kb;
      v16bf Bh = CAT16(*(const v8bf*)wph, *(const v8bf*)(wph+16));
      v16bf Bl = CAT16(*(const v8bf*)wpl, *(const v8bf*)(wpl+16));
      acc[n] = WMMA_BF16(Ah, Bh, acc[n]);
      acc[n] = WMMA_BF16(Ah, Bl, acc[n]);
      acc[n] = WMMA_BF16(Al, Bh, acc[n]);
    }
  }
  int rbase = mt*16 + ((lane < 16) ? 0 : 8);
  #pragma unroll
  for (int n = 0; n < 4; ++n){
    int col = (ntg*4+n)*16 + ln;
    #pragma unroll
    for (int r = 0; r < 8; ++r)
      out[(size_t)(rbase + r)*GATES + col] = acc[n][r];
  }
}

// ---------------------------------------------------------------- K2: per-t BN stats
__global__ void __launch_bounds__(256) aih_stats_kernel(
    const float* __restrict__ aih, float* __restrict__ stats){
  int t = blockIdx.x;
  const float* p = aih + (size_t)t * (BATCH*GATES);
  float s = 0.f, q = 0.f;
  for (int i = threadIdx.x; i < BATCH*GATES; i += 256){
    float v = p[i]; s += v; q += v*v;
  }
  #pragma unroll
  for (int m = 1; m < 32; m <<= 1){
    s += __shfl_xor(s, m, 32);
    q += __shfl_xor(q, m, 32);
  }
  __shared__ float rs[8], rq[8];
  int wv = threadIdx.x >> 5;
  if ((threadIdx.x & 31) == 0){ rs[wv] = s; rq[wv] = q; }
  __syncthreads();
  if (threadIdx.x == 0){
    float S = 0.f, Q = 0.f;
    #pragma unroll
    for (int i = 0; i < 8; ++i){ S += rs[i]; Q += rq[i]; }
    float mn = S / (float)(BATCH*GATES);
    float vr = Q / (float)(BATCH*GATES) - mn*mn;
    vr = fmaxf(vr, 0.f);
    stats[2*t]   = mn;
    stats[2*t+1] = rsqrtf(vr + EPS);
  }
}

// ---------------------------------------------------------------- K3: recurrence
// 1 block, 1024 threads = 32 waves. Wave w: rows m0 = (w>=16)?16:0,
// column group q = w&15 owns gate columns {g*256 + q*16 .. +15, g=0..3},
// so i/f/g/o for hidden unit j = q*16+l live in this wave's registers.
__global__ void __launch_bounds__(1024) bnlstm_seq_kernel(
    const bf16_t* __restrict__ Whh_h, const bf16_t* __restrict__ Whh_l,
    const float*  __restrict__ aih,   const float* __restrict__ stats,
    const float*  __restrict__ bias_ih, const float* __restrict__ bias_hh,
    const float*  __restrict__ bn0_w, const float* __restrict__ bn0_b,
    const float*  __restrict__ bn1_w, const float* __restrict__ bn1_b,
    const float*  __restrict__ bn2_w, const float* __restrict__ bn2_b,
    float* __restrict__ out){
  const int tid  = threadIdx.x;
  const int wv   = tid >> 5;             // 0..31
  const int lane = tid & 31;
  const int q    = wv & 15;              // column group
  const int m0   = (wv >> 4) * 16;       // batch-row tile base (0 or 16)
  const int ln   = lane & 15;
  const int khalf = (lane < 16) ? 0 : 8;
  const int rsel  = (lane < 16) ? 0 : 8; // C/D fragment row selector
  const int cj    = q*16 + ln;           // hidden unit owned by this lane

  __shared__ __align__(16) bf16_t h_hi[BATCH*HID];
  __shared__ __align__(16) bf16_t h_lo[BATCH*HID];
  __shared__ float red[64];
  __shared__ float bcast[2];
  __shared__ float colsum[HID], colsumsq[HID], col_m[HID], col_r[HID];

  for (int i = tid; i < BATCH*HID; i += 1024){
    h_hi[i] = (bf16_t)0.0f; h_lo[i] = (bf16_t)0.0f;
  }

  // fold bn0_b + bn1_b + biases into one per-gate constant
  const float s0 = bn0_w[0], s1 = bn1_w[0];
  const float bfold = bn0_b[0] + bn1_b[0];
  float biasc[4];
  #pragma unroll
  for (int g = 0; g < 4; ++g){
    int col = g*256 + cj;
    biasc[g] = bias_ih[col] + bias_hh[col] + bfold;
  }
  const float w2 = bn2_w[cj], b2 = bn2_b[cj];

  // warm the cache for step 0's gate inputs
  __builtin_prefetch(aih + tid*32, 0, 1);

  v8f c = {};                 // cell state fragment, lives in registers
  __syncthreads();

  #pragma unroll 1
  for (int t = 0; t < T_STEPS; ++t){
    // prefetch next step's a_ih slice (128KB, one cacheline per thread)
    // while this step's WMMA chain executes -> gate loads hit cache.
    {
      int tn = (t + 1 < T_STEPS) ? (t + 1) : t;
      __builtin_prefetch(aih + (size_t)tn * (BATCH*GATES) + tid*32, 0, 1);
    }

    // ---- z = h @ Whh^T  (bf16x3 split, 96 WMMAs/wave/step) ----
    v8f acc[4] = {{},{},{},{}};
    #pragma unroll
    for (int ks = 0; ks < 8; ++ks){
      int kb = ks*32 + khalf;
      const bf16_t* hp = h_hi + (m0 + ln)*HID + kb;
      const bf16_t* lp = h_lo + (m0 + ln)*HID + kb;
      v16bf Ah = CAT16(*(const v8bf*)hp, *(const v8bf*)(hp+16));
      v16bf Al = CAT16(*(const v8bf*)lp, *(const v8bf*)(lp+16));
      #pragma unroll
      for (int g = 0; g < 4; ++g){
        const bf16_t* wph = Whh_h + (size_t)(g*256 + q*16 + ln)*HID + kb;
        const bf16_t* wpl = Whh_l + (size_t)(g*256 + q*16 + ln)*HID + kb;
        v16bf Bh = CAT16(*(const v8bf*)wph, *(const v8bf*)(wph+16));
        v16bf Bl = CAT16(*(const v8bf*)wpl, *(const v8bf*)(wpl+16));
        acc[g] = WMMA_BF16(Ah, Bh, acc[g]);
        acc[g] = WMMA_BF16(Ah, Bl, acc[g]);
        acc[g] = WMMA_BF16(Al, Bh, acc[g]);
      }
    }

    // ---- scalar-BN stats over all 32768 z values ----
    float s = 0.f, qq = 0.f;
    #pragma unroll
    for (int g = 0; g < 4; ++g)
      #pragma unroll
      for (int r = 0; r < 8; ++r){ float v = acc[g][r]; s += v; qq += v*v; }
    #pragma unroll
    for (int msk = 1; msk < 32; msk <<= 1){
      s  += __shfl_xor(s,  msk, 32);
      qq += __shfl_xor(qq, msk, 32);
    }
    if (lane == 0){ red[wv] = s; red[32+wv] = qq; }
    if (tid < HID){ colsum[tid] = 0.f; colsumsq[tid] = 0.f; }
    __syncthreads();
    if (tid == 0){
      float S = 0.f, Q = 0.f;
      #pragma unroll
      for (int i = 0; i < 32; ++i){ S += red[i]; Q += red[32+i]; }
      float mn = S / (float)(BATCH*GATES);
      float vr = fmaxf(Q / (float)(BATCH*GATES) - mn*mn, 0.f);
      bcast[0] = mn; bcast[1] = rsqrtf(vr + EPS);
    }
    __syncthreads();
    const float mh = bcast[0], rh = bcast[1];
    const float mi = stats[2*t], ri = stats[2*t+1];

    // ---- gates + cell update, all in-register ----
    const float* ap = aih + (size_t)t * (BATCH*GATES);
    float cs = 0.f, cq = 0.f;
    #pragma unroll
    for (int r = 0; r < 8; ++r){
      int row = m0 + rsel + r;
      const float* arow = ap + (size_t)row*GATES;
      float gi = (acc[0][r]-mh)*rh*s0 + (arow[      cj]-mi)*ri*s1 + biasc[0];
      float gf = (acc[1][r]-mh)*rh*s0 + (arow[256 + cj]-mi)*ri*s1 + biasc[1];
      float gg = (acc[2][r]-mh)*rh*s0 + (arow[512 + cj]-mi)*ri*s1 + biasc[2];
      float go = (acc[3][r]-mh)*rh*s0 + (arow[768 + cj]-mi)*ri*s1 + biasc[3];
      float cv = fsigmoid(gf)*c[r] + fsigmoid(gi)*ftanh(gg);
      c[r] = cv;
      acc[3][r] = go;           // keep o-gate preactivation for after BN2
      cs += cv; cq += cv*cv;
    }
    // ---- channel-BN stats over batch (ds_add_f32 atomics) ----
    atomicAdd(&colsum[cj],  cs);
    atomicAdd(&colsumsq[cj], cq);
    __syncthreads();
    if (tid < HID){
      float mn = colsum[tid] / (float)BATCH;
      float vr = fmaxf(colsumsq[tid] / (float)BATCH - mn*mn, 0.f);
      col_m[tid] = mn; col_r[tid] = rsqrtf(vr + EPS);
    }
    __syncthreads();
    const float m2 = col_m[cj], r2 = col_r[cj];

    // ---- h_new: global out + bf16 hi/lo back to LDS ----
    float* op = out + (size_t)t * (BATCH*HID);
    #pragma unroll
    for (int r = 0; r < 8; ++r){
      int row = m0 + rsel + r;
      float hn = fsigmoid(acc[3][r]) * ftanh((c[r]-m2)*r2*w2 + b2);
      op[row*HID + cj] = hn;
      bf16_t hh = (bf16_t)hn;
      h_hi[row*HID + cj] = hh;
      h_lo[row*HID + cj] = (bf16_t)(hn - (float)hh);
    }
    __syncthreads();   // protect h_hi/h_lo for next step's A loads
  }
}

// ---------------------------------------------------------------- launcher
extern "C" void kernel_launch(void* const* d_in, const int* in_sizes, int n_in,
                              void* d_out, int out_size, void* d_ws, size_t ws_size,
                              hipStream_t stream) {
  const float* seq     = (const float*)d_in[0];   // [2048,32,256]
  const float* w_ih    = (const float*)d_in[1];   // [1024,256]
  const float* w_hh    = (const float*)d_in[2];   // [1024,256]
  const float* bias_ih = (const float*)d_in[3];
  const float* bias_hh = (const float*)d_in[4];
  const float* bn0_w   = (const float*)d_in[5];
  const float* bn0_b   = (const float*)d_in[6];
  const float* bn1_w   = (const float*)d_in[7];
  const float* bn1_b   = (const float*)d_in[8];
  const float* bn2_w   = (const float*)d_in[9];
  const float* bn2_b   = (const float*)d_in[10];
  float* out = (float*)d_out;

  const size_t NX = (size_t)T_STEPS*BATCH*HID;    // 16,777,216
  const size_t NW = (size_t)GATES*HID;            // 262,144

  bf16_t* xhi   = (bf16_t*)d_ws;
  bf16_t* xlo   = xhi   + NX;
  bf16_t* wihhi = xlo   + NX;
  bf16_t* wihlo = wihhi + NW;
  bf16_t* whhhi = wihlo + NW;
  bf16_t* whhlo = whhhi + NW;
  float*  aih   = (float*)(whhlo + NW);           // [65536,1024] fp32
  float*  stats = aih + NX*4;                     // [2048][2]

  // K0: split-convert inputs/weights to bf16 hi/lo
  cvt_split_kernel<<<(int)(NX+255)/256, 256, 0, stream>>>(seq,  xhi,   xlo,   (int)NX);
  cvt_split_kernel<<<(int)(NW+255)/256, 256, 0, stream>>>(w_ih, wihhi, wihlo, (int)NW);
  cvt_split_kernel<<<(int)(NW+255)/256, 256, 0, stream>>>(w_hh, whhhi, whhlo, (int)NW);

  // K1: bulk GEMM a_ih = X @ W_ih^T  (4096 x 16 strips, 8 waves/block)
  gemm_ih_kernel<<<8192, 256, 0, stream>>>(xhi, xlo, wihhi, wihlo, aih);

  // K2: per-timestep BN stats of a_ih
  aih_stats_kernel<<<T_STEPS, 256, 0, stream>>>(aih, stats);

  // K3: persistent recurrent workgroup
  bnlstm_seq_kernel<<<1, 1024, 0, stream>>>(whhhi, whhlo, aih, stats,
                                            bias_ih, bias_hh,
                                            bn0_w, bn0_b, bn1_w, bn1_b,
                                            bn2_w, bn2_b, out);
}
